// SCFA_65635690217598
// MI455X (gfx1250) — matrix operand order repaired and checked by reading, not compiled, not run on hardware
//
#include <hip/hip_runtime.h>
#include <math.h>

#define B_      8
#define C_      192
#define C2_     384
#define HW_     16384
#define HEADS_  6
#define HD_     32
#define KSPLIT_ 32

typedef _Float16 half_t;
typedef __attribute__((ext_vector_type(16))) _Float16 v16h;
typedef __attribute__((ext_vector_type(8)))  _Float16 v8h;
typedef __attribute__((ext_vector_type(8)))  float    v8f;

__device__ __forceinline__ v8f wmma_f16(v16h a, v16h b, v8f c) {
  // D = A(16x32 f16) x B(32x16 f16) + C(16x16 f32)
  return __builtin_amdgcn_wmma_f32_16x16x32_f16(false, a, false, b, (short)0, c, false, false);
}

// A-fragment loader per ISA 7.12.2 (16-bit A 16x32): lane holds K = ksel+0..7 and ksel+16..23
__device__ __forceinline__ v16h pack_a(const half_t* p) {
  v8h lo = *(const v8h*)(p);
  v8h hi = *(const v8h*)(p + 16);
  v16h r;
#pragma unroll
  for (int i = 0; i < 8; i++) { r[i] = lo[i]; r[i + 8] = hi[i]; }
  return r;
}

// ---------------- K1: weights f32 -> f16 ----------------
__global__ void k_prep_w(const float* __restrict__ wq, const float* __restrict__ wk,
                         half_t* __restrict__ Wh, half_t* __restrict__ Wl) {
  int i = blockIdx.x * 256 + threadIdx.x;
  if (i < C2_ * C_) { Wh[i] = (half_t)wq[i]; Wl[i] = (half_t)wk[i]; }
}

// ---------------- K2: transpose-convert X f32 [c,s] -> Xt f16 [s,c] ----------------
__global__ void k_transpose(const float* __restrict__ hsi, const float* __restrict__ lidar,
                            half_t* __restrict__ Xth, half_t* __restrict__ Xtl) {
  __shared__ half_t tile[32 * C_];
  int b = blockIdx.y, mod = blockIdx.z;
  const float* x = (mod ? lidar : hsi) + (size_t)b * C_ * HW_;
  half_t* xt = (mod ? Xtl : Xth) + (size_t)b * HW_ * C_;
  int s0 = blockIdx.x * 32;
  for (int idx = threadIdx.x; idx < 32 * C_; idx += 256) {
    int j = idx & 31, c = idx >> 5;
    tile[j * C_ + c] = (half_t)x[(size_t)c * HW_ + s0 + j];
  }
  __syncthreads();
  for (int idx = threadIdx.x; idx < 32 * C_; idx += 256) {
    int j = idx / C_, c = idx - j * C_;
    xt[(size_t)(s0 + j) * C_ + c] = tile[j * C_ + c];
  }
}

// ---------------- K3: conv1x1 GEMM via WMMA: Y[o,s] = sum_c W[o,c] Xt[s,c] + bias ----------------
// grid: x = HW/512, y = 24 (M tiles), z = b*2+mod. block = 256 (8 waves), wave = 16x64 strip.
__global__ void k_conv1x1(const half_t* __restrict__ Wh, const half_t* __restrict__ Wl,
                          const half_t* __restrict__ Xth, const half_t* __restrict__ Xtl,
                          const float* __restrict__ bh, const float* __restrict__ bl,
                          half_t* __restrict__ Yh, half_t* __restrict__ Yl) {
  const int lane = threadIdx.x & 31;
  const int wave = threadIdx.x >> 5;
  const int z = blockIdx.z, mod = z & 1, b = z >> 1;
  const half_t* Wf  = mod ? Wl : Wh;
  const half_t* Xt  = (mod ? Xtl : Xth) + (size_t)b * HW_ * C_;
  const float*  bias = mod ? bl : bh;
  half_t* Y = (mod ? Yl : Yh) + (size_t)b * C2_ * HW_;

  const int o0 = blockIdx.y * 16;
  const int n0 = blockIdx.x * 512 + wave * 64;

  v8f acc0 = {0,0,0,0,0,0,0,0}, acc1 = acc0, acc2 = acc0, acc3 = acc0;
  const int rowA  = o0 + (lane & 15);
  const int kselA = (lane < 16) ? 0 : 16;
  const int sB    = n0 + (lane & 15);
  const int kselB = (lane < 16) ? 0 : 16;

#pragma unroll
  for (int k0 = 0; k0 < C_; k0 += 32) {
    v16h a  = *(const v16h*)(Wf + (size_t)rowA * C_ + k0 + kselA);
    v16h b0 = *(const v16h*)(Xt + (size_t)(sB     ) * C_ + k0 + kselB);
    v16h b1 = *(const v16h*)(Xt + (size_t)(sB + 16) * C_ + k0 + kselB);
    v16h b2 = *(const v16h*)(Xt + (size_t)(sB + 32) * C_ + k0 + kselB);
    v16h b3 = *(const v16h*)(Xt + (size_t)(sB + 48) * C_ + k0 + kselB);
    acc0 = wmma_f16(a, b0, acc0);
    acc1 = wmma_f16(a, b1, acc1);
    acc2 = wmma_f16(a, b2, acc2);
    acc3 = wmma_f16(a, b3, acc3);
  }
  // D layout: VGPR v -> M = v + 8*(lane>=16); N = lane&15
#pragma unroll
  for (int v = 0; v < 8; v++) {
    int o = o0 + v + ((lane >> 4) << 3);
    float bv = bias[o];
    size_t base = (size_t)o * HW_ + n0 + (lane & 15);
    Y[base     ] = (half_t)(acc0[v] + bv);
    Y[base + 16] = (half_t)(acc1[v] + bv);
    Y[base + 32] = (half_t)(acc2[v] + bv);
    Y[base + 48] = (half_t)(acc3[v] + bv);
  }
}

// ---------------- K4: depthwise 3x3 SAME + bias, f16->f16 ----------------
__global__ void k_dwconv(const half_t* __restrict__ Ylin_h, const half_t* __restrict__ Ylin_l,
                         const float* __restrict__ dwh, const float* __restrict__ dwl,
                         const float* __restrict__ dbh, const float* __restrict__ dbl,
                         half_t* __restrict__ QVh, half_t* __restrict__ QVl) {
  int ch = blockIdx.y, z = blockIdx.z, mod = z & 1, b = z >> 1;
  const half_t* in  = (mod ? Ylin_l : Ylin_h) + ((size_t)b * C2_ + ch) * HW_;
  const float*  wz  = (mod ? dwl : dwh) + ch * 9;
  float bias        = (mod ? dbl : dbh)[ch];
  half_t* outp      = (mod ? QVl : QVh) + ((size_t)b * C2_ + ch) * HW_;
  int s = blockIdx.x * 256 + threadIdx.x;
  int y = s >> 7, x = s & 127;
  float acc = bias;
#pragma unroll
  for (int dy = -1; dy <= 1; dy++) {
    int yy = y + dy;
    if ((unsigned)yy >= 128u) continue;
#pragma unroll
    for (int dx = -1; dx <= 1; dx++) {
      int xx = x + dx;
      if ((unsigned)xx >= 128u) continue;
      acc += wz[(dy + 1) * 3 + (dx + 1)] * (float)in[yy * 128 + xx];
    }
  }
  outp[s] = (half_t)acc;
}

// ---------------- K5: row sum-of-squares -> 1/max(||.||,eps) ----------------
// grid (C_, B_, 2): z=0 -> q rows (hsi QV ch 0..191), z=1 -> k rows (lidar QV ch 0..191)
__global__ void k_norms(const half_t* __restrict__ QVh, const half_t* __restrict__ QVl,
                        float* __restrict__ inv_nq, float* __restrict__ inv_nk) {
  int c = blockIdx.x, b = blockIdx.y, which = blockIdx.z;
  const half_t* src = ((which == 0) ? QVh : QVl) + ((size_t)b * C2_ + c) * HW_;
  float ss = 0.f;
  for (int i = threadIdx.x; i < HW_; i += 256) { float v = (float)src[i]; ss += v * v; }
  __shared__ float red[256];
  red[threadIdx.x] = ss;
  __syncthreads();
  for (int off = 128; off > 0; off >>= 1) {
    if (threadIdx.x < (unsigned)off) red[threadIdx.x] += red[threadIdx.x + off];
    __syncthreads();
  }
  if (threadIdx.x == 0) {
    float n = fmaxf(sqrtf(red[0]), 1e-12f);
    (which ? inv_nk : inv_nq)[b * C_ + c] = 1.f / n;
  }
}

// ---------------- K6: attn logits (unnormalized) via WMMA, K-split partials ----------------
// grid (48, KSPLIT_), block 32 (one wave). part[bh][split][32][32] f32.
__global__ void k_attn_gemm(const half_t* __restrict__ QVh, const half_t* __restrict__ QVl,
                            float* __restrict__ part) {
  const int lane = threadIdx.x;
  const int bh = blockIdx.x, split = blockIdx.y;
  const int b = bh / HEADS_, h = bh % HEADS_;
  const half_t* q = QVh + ((size_t)b * C2_ + h * HD_) * HW_;
  const half_t* k = QVl + ((size_t)b * C2_ + h * HD_) * HW_;

  v8f a00 = {0,0,0,0,0,0,0,0}, a01 = a00, a10 = a00, a11 = a00;
  const int r     = lane & 15;
  const int kselA = (lane < 16) ? 0 : 16;
  const int kselB = (lane < 16) ? 0 : 16;
  const int sBeg  = split * (HW_ / KSPLIT_);

  for (int s0 = sBeg; s0 < sBeg + HW_ / KSPLIT_; s0 += 32) {
    v16h aq0 = *(const v16h*)(q + (size_t)(r     ) * HW_ + s0 + kselA);
    v16h aq1 = *(const v16h*)(q + (size_t)(r + 16) * HW_ + s0 + kselA);
    v16h bk0 = *(const v16h*)(k + (size_t)(r     ) * HW_ + s0 + kselB);
    v16h bk1 = *(const v16h*)(k + (size_t)(r + 16) * HW_ + s0 + kselB);
    a00 = wmma_f16(aq0, bk0, a00);
    a01 = wmma_f16(aq0, bk1, a01);
    a10 = wmma_f16(aq1, bk0, a10);
    a11 = wmma_f16(aq1, bk1, a11);
  }
  float* p = part + ((size_t)bh * KSPLIT_ + split) * 1024;
#pragma unroll
  for (int v = 0; v < 8; v++) {
    int c = v + ((lane >> 4) << 3);
    int d = lane & 15;
    p[c * 32 + d]             = a00[v];
    p[c * 32 + d + 16]        = a01[v];
    p[(c + 16) * 32 + d]      = a10[v];
    p[(c + 16) * 32 + d + 16] = a11[v];
  }
}

// ---------------- K7: reduce partials + l2norm scaling + temperature + softmax ----------------
// grid (48), block 1024: one wave per row c (lane = d).
__global__ void k_softmax(const float* __restrict__ part, const float* __restrict__ inv_nq,
                          const float* __restrict__ inv_nk, const float* __restrict__ temp,
                          float* __restrict__ attn) {
  int bh = blockIdx.x, b = bh / HEADS_, h = bh % HEADS_;
  int c = threadIdx.x >> 5, d = threadIdx.x & 31;
  const float* p = part + (size_t)bh * KSPLIT_ * 1024;
  float s = 0.f;
  for (int sp = 0; sp < KSPLIT_; sp++) s += p[sp * 1024 + c * 32 + d];
  float l = s * inv_nq[b * C_ + h * HD_ + c] * inv_nk[b * C_ + h * HD_ + d] * temp[h];
  float m = l;
  for (int off = 16; off > 0; off >>= 1) m = fmaxf(m, __shfl_xor(m, off, 32));
  float e = expf(l - m);
  float sum = e;
  for (int off = 16; off > 0; off >>= 1) sum += __shfl_xor(sum, off, 32);
  attn[(size_t)bh * 1024 + c * 32 + d] = e / sum;
}

// ---------------- K8: out = attn@v + v + x for both modalities ----------------
// grid (HW/256, 48), block 256; each thread owns one spatial s for one (b,h).
__global__ void k_output(const float* __restrict__ attn,
                         const half_t* __restrict__ QVh, const half_t* __restrict__ QVl,
                         const float* __restrict__ hsi, const float* __restrict__ lidar,
                         float* __restrict__ out) {
  __shared__ float satt[1024];
  int bh = blockIdx.y, b = bh / HEADS_, h = bh % HEADS_;
  for (int i = threadIdx.x; i < 1024; i += 256) satt[i] = attn[(size_t)bh * 1024 + i];
  __syncthreads();
  int s = blockIdx.x * 256 + threadIdx.x;
  const half_t* vh = QVh + ((size_t)b * C2_ + C_ + h * HD_) * HW_ + s;
  const half_t* vl = QVl + ((size_t)b * C2_ + C_ + h * HD_) * HW_ + s;
  float oh[HD_], ol[HD_];
#pragma unroll
  for (int c = 0; c < HD_; c++) { oh[c] = 0.f; ol[c] = 0.f; }
  for (int d = 0; d < HD_; d++) {
    float vhd = (float)vh[(size_t)d * HW_];
    float vld = (float)vl[(size_t)d * HW_];
    oh[d] += vhd;   // + v term
    ol[d] += vld;
#pragma unroll
    for (int c = 0; c < HD_; c++) {
      float w = satt[c * 32 + d];
      oh[c] += w * vhd;
      ol[c] += w * vld;
    }
  }
  const float* xh = hsi   + ((size_t)b * C_ + h * HD_) * HW_ + s;
  const float* xl = lidar + ((size_t)b * C_ + h * HD_) * HW_ + s;
  float* po_h = out + ((size_t)b * C_ + h * HD_) * HW_ + s;
  float* po_l = out + (size_t)B_ * C_ * HW_ + ((size_t)b * C_ + h * HD_) * HW_ + s;
#pragma unroll
  for (int c = 0; c < HD_; c++) {
    po_h[(size_t)c * HW_] = oh[c] + xh[(size_t)c * HW_];
    po_l[(size_t)c * HW_] = ol[c] + xl[(size_t)c * HW_];
  }
}

extern "C" void kernel_launch(void* const* d_in, const int* in_sizes, int n_in,
                              void* d_out, int out_size, void* d_ws, size_t ws_size,
                              hipStream_t stream) {
  const float* hsi   = (const float*)d_in[0];
  const float* lidar = (const float*)d_in[1];
  const float* qv_w  = (const float*)d_in[2];
  const float* qv_b  = (const float*)d_in[3];
  const float* kv_w  = (const float*)d_in[4];
  const float* kv_b  = (const float*)d_in[5];
  const float* dw_h  = (const float*)d_in[6];
  const float* db_h  = (const float*)d_in[7];
  const float* dw_l  = (const float*)d_in[8];
  const float* db_l  = (const float*)d_in[9];
  const float* temp  = (const float*)d_in[10];
  float* out = (float*)d_out;

  char* p = (char*)d_ws;
  auto carve = [&](size_t bytes) -> char* {
    char* r = p;
    p += (bytes + 255) & ~(size_t)255;
    return r;
  };
  half_t* Wh      = (half_t*)carve((size_t)C2_ * C_ * 2);
  half_t* Wl      = (half_t*)carve((size_t)C2_ * C_ * 2);
  half_t* Xth     = (half_t*)carve((size_t)B_ * HW_ * C_ * 2);
  half_t* Xtl     = (half_t*)carve((size_t)B_ * HW_ * C_ * 2);
  half_t* Ylin_h  = (half_t*)carve((size_t)B_ * C2_ * HW_ * 2);
  half_t* Ylin_l  = (half_t*)carve((size_t)B_ * C2_ * HW_ * 2);
  half_t* QVh     = (half_t*)carve((size_t)B_ * C2_ * HW_ * 2);
  half_t* QVl     = (half_t*)carve((size_t)B_ * C2_ * HW_ * 2);
  float*  inv_nq  = (float*)carve((size_t)B_ * C_ * 4);
  float*  inv_nk  = (float*)carve((size_t)B_ * C_ * 4);
  float*  part    = (float*)carve((size_t)48 * KSPLIT_ * 1024 * 4);
  float*  attn    = (float*)carve((size_t)48 * 1024 * 4);

  // 1) weights -> f16
  k_prep_w<<<dim3((C2_ * C_ + 255) / 256), dim3(256), 0, stream>>>(qv_w, kv_w, Wh, Wl);
  // 2) transpose-convert inputs -> [s,c] f16
  k_transpose<<<dim3(HW_ / 32, B_, 2), dim3(256), 0, stream>>>(hsi, lidar, Xth, Xtl);
  // 3) conv1x1 via WMMA
  k_conv1x1<<<dim3(HW_ / 512, C2_ / 16, B_ * 2), dim3(256), 0, stream>>>(
      Wh, Wl, Xth, Xtl, qv_b, kv_b, Ylin_h, Ylin_l);
  // 4) depthwise 3x3 + bias
  k_dwconv<<<dim3(HW_ / 256, C2_, B_ * 2), dim3(256), 0, stream>>>(
      Ylin_h, Ylin_l, dw_h, dw_l, db_h, db_l, QVh, QVl);
  // 5) l2 norms of q (hsi) and k (lidar) rows
  k_norms<<<dim3(C_, B_, 2), dim3(256), 0, stream>>>(QVh, QVl, inv_nq, inv_nk);
  // 6) q.k^T partials via WMMA
  k_attn_gemm<<<dim3(B_ * HEADS_, KSPLIT_), dim3(32), 0, stream>>>(QVh, QVl, part);
  // 7) reduce + scale + softmax
  k_softmax<<<dim3(B_ * HEADS_), dim3(1024), 0, stream>>>(part, inv_nq, inv_nk, temp, attn);
  // 8) attn@v + v + x for both outputs
  k_output<<<dim3(HW_ / 256, B_ * HEADS_), dim3(256), 0, stream>>>(
      attn, QVh, QVl, hsi, lidar, out);
}
